// GptOssDecoderLayer_52536039965402
// MI455X (gfx1250) — matrix operand order repaired and checked by reading, not compiled, not run on hardware
//
#include <hip/hip_runtime.h>
#include <hip/hip_bf16.h>
#include <math.h>

#define T_TOK 1024
#define H_DIM 2880
#define NHEAD 64
#define NKVH  8
#define D_HEAD 64
#define E_EXP 32
#define K_TOP 4
#define I_DIM 1440
#define QKV_DIM 5120          // (NHEAD + 2*NKVH) * D_HEAD
#define O_DIM  4096           // NHEAD * D_HEAD
#define EPS_V  1e-5f
#define LIMITV 7.0f
#define ALPHAV 1.702f
#define THETAV 150000.0f

typedef __bf16 bf16;
typedef __attribute__((ext_vector_type(16))) __bf16 bf16x16;
typedef __attribute__((ext_vector_type(8)))  float  f32x8;
typedef __attribute__((ext_vector_type(4)))  unsigned u32x4;
typedef __attribute__((ext_vector_type(8)))  int      i32x8;
typedef __attribute__((ext_vector_type(4)))  int      i32x4;

__device__ __forceinline__ int lane_id() { return (int)(threadIdx.x & 31u); }

// ---- CDNA5 async global->LDS copy (ASYNCcnt-tracked, bypasses VGPRs) ------
__device__ __forceinline__ void async_copy_b128(void* lds_dst, const void* gsrc) {
  unsigned loff = (unsigned)(size_t)lds_dst;
  unsigned long long ga = (unsigned long long)(size_t)gsrc;
  asm volatile("global_load_async_to_lds_b128 %0, %1, off"
               :: "v"(loff), "v"(ga) : "memory");
}
__device__ __forceinline__ void wait_async0() {
  asm volatile("s_wait_asynccnt 0x0" ::: "memory");
}

// ---- CDNA5 Tensor Data Mover: dense 2D bf16 tile -> LDS (TENSORcnt) -------
// D# per ISA 8.3/8.4: group0 = {flags, lds_addr, global_addr, type=2};
// group1 = {mask/size/pad, tensor dims, tile dims, dim0 stride}. data_size=2B.
// pad_en inserts (pad_amount+1) DWORDs every (2<<pad_interval) DWORDs into LDS,
// which produces the padded LDS row stride in hardware. OOB rows read as 0.
__device__ __forceinline__ void tdm_load_2d_bf16(
    void* lds_dst, const void* gsrc,
    unsigned tile_d0, unsigned tile_d1,        // elements per row, #rows
    unsigned tensor_d0, unsigned tensor_d1,    // extents for OOB zero-fill
    unsigned stride_d0,                        // row stride, elements
    unsigned pad_en, unsigned pad_interval, unsigned pad_amount) {
  unsigned long long ga = (unsigned long long)(size_t)gsrc;
  u32x4 g0;
  g0[0] = 1u;                                             // count=1 (valid D#)
  g0[1] = (unsigned)(size_t)lds_dst;                      // lds_addr
  g0[2] = (unsigned)(ga & 0xFFFFFFFFu);                   // global_addr[31:0]
  g0[3] = (unsigned)((ga >> 32) & 0x1FFFFFFu) | (2u << 30); // [56:32] | type=2
  i32x8 g1;
  g1[0] = (int)((1u << 16) | (pad_en << 20) |
                (pad_interval << 22) | (pad_amount << 25)); // data_size=2B
  g1[1] = (int)((tensor_d0 & 0xFFFFu) << 16);             // dim0[15:0]
  g1[2] = (int)(((tensor_d0 >> 16) & 0xFFFFu) | ((tensor_d1 & 0xFFFFu) << 16));
  g1[3] = (int)(((tensor_d1 >> 16) & 0xFFFFu) | (tile_d0 << 16));
  g1[4] = (int)(tile_d1 & 0xFFFFu);                       // tile_dim1, dim2=0
  g1[5] = (int)stride_d0;                                 // dim0_stride[31:0]
  g1[6] = 0;
  g1[7] = 0;
  i32x4 z4 = {0, 0, 0, 0};
#if __clang_major__ >= 23
  i32x8 z8 = {0, 0, 0, 0, 0, 0, 0, 0};
  __builtin_amdgcn_tensor_load_to_lds(g0, g1, z4, z4, z8, 0);
#else
  __builtin_amdgcn_tensor_load_to_lds(g0, g1, z4, z4, 0);
#endif
}
__device__ __forceinline__ void wait_tensor0() {
  __builtin_amdgcn_s_wait_tensorcnt(0);
}

// ---- WMMA fragment helpers (CDNA5 wave32 layouts, 16x16x32 bf16) ----------
__device__ __forceinline__ bf16x16 frag_a_ld(const bf16* src, int ld) {
  int l = lane_id(); int row = l & 15; int g = l >> 4;
  bf16x16 a;
#pragma unroll
  for (int e = 0; e < 16; ++e) {
    int k = g * 16 + e;
    a[e] = src[row * ld + k];
  }
  return a;
}

__device__ __forceinline__ bf16x16 frag_b_wt(const bf16* src, int ld) {
  int l = lane_id(); int n = l & 15; int g = l >> 4;
  bf16x16 b;
#pragma unroll
  for (int e = 0; e < 16; ++e) b[e] = src[n * ld + g * 16 + e];
  return b;
}

__device__ __forceinline__ bf16x16 frag_b_nt(const bf16* src, int ld) {
  int l = lane_id(); int n = l & 15; int g = l >> 4;
  bf16x16 b;
#pragma unroll
  for (int e = 0; e < 16; ++e) b[e] = src[(g * 16 + e) * ld + n];
  return b;
}

__device__ __forceinline__ f32x8 wmma_bf16(bf16x16 a, bf16x16 b, f32x8 c) {
  return __builtin_amdgcn_wmma_f32_16x16x32_bf16(
      false, a, false, b, (short)0, c, false, false);
}

// ---- RMSNorm: fp32 in, bf16 out -------------------------------------------
__global__ __launch_bounds__(256) void rmsnorm_kernel(
    const float* __restrict__ x, const float* __restrict__ w,
    bf16* __restrict__ out) {
  __shared__ float red[256];
  int t = blockIdx.x;
  const float* row = x + (size_t)t * H_DIM;
  float s = 0.f;
  for (int i = threadIdx.x; i < H_DIM; i += 256) { float v = row[i]; s += v * v; }
  red[threadIdx.x] = s;
  __syncthreads();
  for (int off = 128; off > 0; off >>= 1) {
    if ((int)threadIdx.x < off) red[threadIdx.x] += red[threadIdx.x + off];
    __syncthreads();
  }
  float rinv = rsqrtf(red[0] / (float)H_DIM + EPS_V);
  for (int i = threadIdx.x; i < H_DIM; i += 256)
    out[(size_t)t * H_DIM + i] = (bf16)(row[i] * rinv * w[i]);
}

// ---- C[M,N] = A[M,K]*W[N,K]^T (+bias +resid): 128x64 tile, 4 WMMA/wave ----
// A tile staged by TDM (wave 0 issues one descriptor per k-step; hardware
// inserts the 16B LDS row pad: 64B data + 16B pad = 40-element stride).
__global__ __launch_bounds__(256) void gemm_xwT_kernel(
    const bf16* __restrict__ A, const float* __restrict__ W,
    const float* __restrict__ bias, const float* __restrict__ resid,
    float* __restrict__ C, int M, int N, int Kd) {
  __shared__ alignas(16) bf16 As[128][40];
  __shared__ alignas(16) bf16 Bs[64][40];
  int tid = threadIdx.x;
  int m0 = blockIdx.y * 128;
  int n0 = blockIdx.x * 64;
  int w = tid >> 5;
  int mw = w * 16;
  f32x8 cacc[4] = {};
  for (int k0 = 0; k0 < Kd; k0 += 32) {
    if (tid < 32) {                                  // wave 0: TDM A tile
      tdm_load_2d_bf16(&As[0][0], A + (size_t)m0 * Kd + k0,
                       /*tile*/ 32, 128, /*tensor*/ (unsigned)Kd, 128,
                       /*stride*/ (unsigned)Kd,
                       /*pad: 16B every 64B*/ 1, 3, 3);
    }
#pragma unroll
    for (int j = 0; j < 8; ++j) {                    // W fp32 -> bf16 stage
      int idx = tid + j * 256;
      int n = idx >> 5, k = idx & 31;
      Bs[n][k] = (bf16)W[(size_t)(n0 + n) * Kd + k0 + k];
    }
    __builtin_prefetch(&W[(size_t)n0 * Kd + k0 + 128], 0, 1);
    wait_tensor0();
    __syncthreads();
    bf16x16 a = frag_a_ld(&As[mw][0], 40);
#pragma unroll
    for (int nn = 0; nn < 4; ++nn)
      cacc[nn] = wmma_bf16(a, frag_b_wt(&Bs[nn * 16][0], 40), cacc[nn]);
    __syncthreads();
  }
  int l = lane_id(); int nl = l & 15; int g = l >> 4;
#pragma unroll
  for (int nn = 0; nn < 4; ++nn)
#pragma unroll
    for (int r = 0; r < 8; ++r) {
      int m = m0 + mw + g * 8 + r;
      int n = n0 + nn * 16 + nl;
      float v = cacc[nn][r];
      if (bias)  v += bias[n];
      if (resid) v += resid[(size_t)m * N + n];
      C[(size_t)m * N + n] = v;
    }
}

// ---- RoPE + layout change to head-major bf16 Q/K/V ------------------------
__global__ __launch_bounds__(256) void rope_kernel(
    const float* __restrict__ qkv, const int* __restrict__ positions,
    bf16* __restrict__ qbf, bf16* __restrict__ kbf, bf16* __restrict__ vbf) {
  int t = blockIdx.x;
  float pos = (float)positions[t];
  const float* src = qkv + (size_t)t * QKV_DIM;
  const float ln_theta = logf(THETAV);
  for (int idx = threadIdx.x; idx < NHEAD * 32; idx += 256) {   // Q
    int h = idx >> 5, i = idx & 31;
    float ang = pos * expf(-ln_theta * (float)i / 32.f);
    float cs = cosf(ang), sn = sinf(ang);
    float x1 = src[h * 64 + i], x2 = src[h * 64 + i + 32];
    bf16* q = qbf + ((size_t)h * T_TOK + t) * 64;
    q[i]      = (bf16)(x1 * cs - x2 * sn);
    q[i + 32] = (bf16)(x2 * cs + x1 * sn);
  }
  for (int idx = threadIdx.x; idx < NKVH * 32; idx += 256) {    // K
    int h = idx >> 5, i = idx & 31;
    float ang = pos * expf(-ln_theta * (float)i / 32.f);
    float cs = cosf(ang), sn = sinf(ang);
    const float* ks = src + NHEAD * 64;
    float x1 = ks[h * 64 + i], x2 = ks[h * 64 + i + 32];
    bf16* k = kbf + ((size_t)h * T_TOK + t) * 64;
    k[i]      = (bf16)(x1 * cs - x2 * sn);
    k[i + 32] = (bf16)(x2 * cs + x1 * sn);
  }
  for (int idx = threadIdx.x; idx < NKVH * 64; idx += 256) {    // V copy
    int h = idx >> 6, d = idx & 63;
    vbf[((size_t)h * T_TOK + t) * 64 + d] =
        (bf16)src[(NHEAD + NKVH) * 64 + h * 64 + d];
  }
}

// ---- Flash attention with sinks: one wave per (16 queries, head) ----------
__global__ __launch_bounds__(32) void attn_kernel(
    const bf16* __restrict__ qbf, const bf16* __restrict__ kbf,
    const bf16* __restrict__ vbf, const float* __restrict__ sinks,
    bf16* __restrict__ attn_out) {
  __shared__ float Sld[16][36];
  __shared__ bf16  Pld[16][40];
  __shared__ float corr_s[16];
  __shared__ float fscale_s[16];
  int q0 = blockIdx.x * 16;
  int h  = blockIdx.y;
  int kvh = h >> 3;                        // NHEAD / NKVH = 8
  int l = lane_id();
  const bf16* qptr = qbf + ((size_t)h * T_TOK + q0) * 64;
  bf16x16 a0 = frag_a_ld(qptr, 64);        // D = 64 -> two K=32 chunks
  bf16x16 a1 = frag_a_ld(qptr + 32, 64);
  f32x8 o[4]; o[0] = {}; o[1] = {}; o[2] = {}; o[3] = {};
  float m_i = -INFINITY, l_i = 0.f;
  const float scale = 0.125f;              // D^-0.5
  int nkb = (q0 + 16 + 31) >> 5;
  int nl = l & 15, g = l >> 4;
  for (int kb = 0; kb < nkb; ++kb) {
    int key0 = kb * 32;
    const bf16* kp = kbf + ((size_t)kvh * T_TOK + key0) * 64;
    f32x8 s0 = {}, s1 = {};
    s0 = wmma_bf16(a0, frag_b_wt(kp, 64), s0);
    s0 = wmma_bf16(a1, frag_b_wt(kp + 32, 64), s0);
    s1 = wmma_bf16(a0, frag_b_wt(kp + 16 * 64, 64), s1);
    s1 = wmma_bf16(a1, frag_b_wt(kp + 16 * 64 + 32, 64), s1);
#pragma unroll
    for (int r = 0; r < 8; ++r) {          // spill scores to LDS
      Sld[g * 8 + r][nl]      = s0[r];
      Sld[g * 8 + r][nl + 16] = s1[r];
    }
    __syncthreads();
    if (l < 16) {                          // online softmax, one row per lane
      int q = q0 + l;
      float mx = m_i;
      float rowv[32];
#pragma unroll
      for (int j = 0; j < 32; ++j) {
        float v = (key0 + j <= q) ? Sld[l][j] * scale : -INFINITY;
        rowv[j] = v;
        mx = fmaxf(mx, v);
      }
      float cr = expf(m_i - mx);
      float sum = 0.f;
#pragma unroll
      for (int j = 0; j < 32; ++j) {
        float p = expf(rowv[j] - mx);
        sum += p;
        Pld[l][j] = (bf16)p;
      }
      l_i = l_i * cr + sum;
      m_i = mx;
      corr_s[l] = cr;
    }
    __syncthreads();
#pragma unroll
    for (int n = 0; n < 4; ++n)            // rescale accumulators
#pragma unroll
      for (int r = 0; r < 8; ++r) o[n][r] *= corr_s[g * 8 + r];
    bf16x16 p = frag_a_ld(&Pld[0][0], 40);
    const bf16* vp = vbf + ((size_t)kvh * T_TOK + key0) * 64;
#pragma unroll
    for (int n = 0; n < 4; ++n)
      o[n] = wmma_bf16(p, frag_b_nt(vp + n * 16, 64), o[n]);
    __syncthreads();
  }
  if (l < 16) {                            // sink term in normalization
    float snk = sinks[h];
    float mf = fmaxf(m_i, snk);
    float denom = l_i * expf(m_i - mf) + expf(snk - mf);
    fscale_s[l] = expf(m_i - mf) / denom;
  }
  __syncthreads();
#pragma unroll
  for (int n = 0; n < 4; ++n)
#pragma unroll
    for (int r = 0; r < 8; ++r) {
      int q = q0 + g * 8 + r;
      attn_out[(size_t)q * O_DIM + h * 64 + n * 16 + nl] =
          (bf16)(o[n][r] * fscale_s[g * 8 + r]);
    }
}

// ---- Router: logits, top-4, softmax, expert counts ------------------------
__global__ __launch_bounds__(32) void router_kernel(
    const bf16* __restrict__ h2, const float* __restrict__ rw,
    const float* __restrict__ rb, int* __restrict__ topi,
    float* __restrict__ topw, int* __restrict__ counts) {
  __shared__ float lg[E_EXP];
  int t = blockIdx.x;
  int e = threadIdx.x;
  const bf16*  hrow = h2 + (size_t)t * H_DIM;
  const float* wrow = rw + (size_t)e * H_DIM;
  float acc = rb[e];
  for (int i = 0; i < H_DIM; ++i) acc += (float)hrow[i] * wrow[i];
  lg[e] = acc;
  __syncthreads();
  if (e == 0) {
    int idx[K_TOP]; float val[K_TOP]; bool used[E_EXP];
    for (int j = 0; j < E_EXP; ++j) used[j] = false;
    for (int k = 0; k < K_TOP; ++k) {
      int best = 0; float bv = -INFINITY;
      for (int j = 0; j < E_EXP; ++j)
        if (!used[j] && lg[j] > bv) { bv = lg[j]; best = j; }
      used[best] = true; idx[k] = best; val[k] = bv;
    }
    float mx = val[0], sum = 0.f, ex[K_TOP];
    for (int k = 0; k < K_TOP; ++k) { ex[k] = expf(val[k] - mx); sum += ex[k]; }
    for (int k = 0; k < K_TOP; ++k) {
      topi[t * K_TOP + k] = idx[k];
      topw[t * K_TOP + k] = ex[k] / sum;
      atomicAdd(&counts[idx[k]], 1);
    }
  }
}

__global__ void moe_scan_kernel(const int* __restrict__ counts,
                                int* __restrict__ offsets, int* __restrict__ fill) {
  if (threadIdx.x == 0) {
    int acc = 0;
    for (int e = 0; e < E_EXP; ++e) { offsets[e] = acc; acc += counts[e]; }
  }
  if ((int)threadIdx.x < E_EXP) fill[threadIdx.x] = 0;
}

__global__ __launch_bounds__(256) void moe_scatter_kernel(
    const int* __restrict__ topi, const float* __restrict__ topw,
    const int* __restrict__ offsets, int* __restrict__ fill,
    int* __restrict__ slot_token, float* __restrict__ slot_w) {
  int t = blockIdx.x * 256 + threadIdx.x;
  if (t >= T_TOK) return;
  for (int k = 0; k < K_TOP; ++k) {
    int e = topi[t * K_TOP + k];
    int p = atomicAdd(&fill[e], 1);
    int s = offsets[e] + p;
    slot_token[s] = t;
    slot_w[s] = topw[t * K_TOP + k];
  }
}

__global__ __launch_bounds__(256) void copy_kernel(
    const float* __restrict__ src, float* __restrict__ dst, int n) {
  int i = blockIdx.x * 256 + threadIdx.x;
  if (i < n) dst[i] = src[i];
}

// ---- MoE gate_up GEMM + clamped SwiGLU epilogue ---------------------------
// A rows are a data-dependent gather -> per-lane async b128 copies.
__global__ __launch_bounds__(256) void moe_gateup_kernel(
    const bf16* __restrict__ h2, const float* __restrict__ wgu,
    const float* __restrict__ bgu, const int* __restrict__ offsets,
    const int* __restrict__ counts, const int* __restrict__ slot_token,
    bf16* __restrict__ act) {
  __shared__ alignas(16) bf16 As[64][40];
  __shared__ alignas(16) bf16 Bs[64][40];
  __shared__ float Ct[64][66];
  int e = blockIdx.z;
  int cnt = counts[e];
  int row0 = blockIdx.y * 64;
  if (row0 >= cnt) return;
  int rows = min(64, cnt - row0);
  int base = offsets[e] + row0;
  int n0 = blockIdx.x * 64;
  const float* W = wgu + (size_t)e * (2 * I_DIM) * H_DIM;
  int tid = threadIdx.x;
  int w = tid >> 5, mw = (w >> 1) * 16, nw = (w & 1) * 32;
  f32x8 c0 = {}, c1 = {};
  for (int k0 = 0; k0 < H_DIM; k0 += 32) {
    {
      int r = tid >> 2, cc = tid & 3;
      if (r < rows) {
        int tok = slot_token[base + r];
        async_copy_b128(&As[r][cc * 8],
                        h2 + (size_t)tok * H_DIM + k0 + cc * 8);
      } else {
#pragma unroll
        for (int q = 0; q < 8; ++q) As[r][cc * 8 + q] = (bf16)0.f;
      }
    }
#pragma unroll
    for (int j = 0; j < 8; ++j) {
      int idx = tid + j * 256;
      int n = idx >> 5, k = idx & 31;
      Bs[n][k] = (bf16)W[(size_t)(n0 + n) * H_DIM + k0 + k];
    }
    __builtin_prefetch(&W[(size_t)n0 * H_DIM + k0 + 128], 0, 1);
    wait_async0();
    __syncthreads();
    bf16x16 a = frag_a_ld(&As[mw][0], 40);
    c0 = wmma_bf16(a, frag_b_wt(&Bs[nw][0], 40), c0);
    c1 = wmma_bf16(a, frag_b_wt(&Bs[nw + 16][0], 40), c1);
    __syncthreads();
  }
  int l = lane_id(), nl = l & 15, g = l >> 4;
#pragma unroll
  for (int r = 0; r < 8; ++r) {
    Ct[mw + g * 8 + r][nw + nl]      = c0[r];
    Ct[mw + g * 8 + r][nw + nl + 16] = c1[r];
  }
  __syncthreads();
#pragma unroll
  for (int j = 0; j < 8; ++j) {            // interleaved gate/up activation
    int idx = tid + j * 256;
    int r = idx >> 5, p = idx & 31;
    if (r < rows) {
      float gv = Ct[r][2 * p]     + bgu[(size_t)e * 2 * I_DIM + n0 + 2 * p];
      float uv = Ct[r][2 * p + 1] + bgu[(size_t)e * 2 * I_DIM + n0 + 2 * p + 1];
      gv = fminf(gv, LIMITV);
      uv = fminf(fmaxf(uv, -LIMITV), LIMITV);
      float sig = 1.f / (1.f + expf(-ALPHAV * gv));
      float av = (uv + 1.f) * (gv * sig);
      act[(size_t)(base + r) * I_DIM + (n0 >> 1) + p] = (bf16)av;
    }
  }
}

// ---- MoE down GEMM + weighted scatter-add into output ---------------------
// A rows are contiguous slots -> dense TDM tile; tensor_dim1 = rows gives
// hardware zero-fill of the tail rows (OOB loads return zero).
__global__ __launch_bounds__(256) void moe_down_kernel(
    const bf16* __restrict__ act, const float* __restrict__ wd,
    const float* __restrict__ bd, const int* __restrict__ offsets,
    const int* __restrict__ counts, const int* __restrict__ slot_token,
    const float* __restrict__ slot_w, float* __restrict__ out) {
  __shared__ alignas(16) bf16 As[64][40];
  __shared__ alignas(16) bf16 Bs[64][40];
  int e = blockIdx.z;
  int cnt = counts[e];
  int row0 = blockIdx.y * 64;
  if (row0 >= cnt) return;
  int rows = min(64, cnt - row0);
  int base = offsets[e] + row0;
  int n0 = blockIdx.x * 64;
  const float* W = wd + (size_t)e * H_DIM * I_DIM;
  int tid = threadIdx.x;
  int w = tid >> 5, mw = (w >> 1) * 16, nw = (w & 1) * 32;
  f32x8 c0 = {}, c1 = {};
  for (int k0 = 0; k0 < I_DIM; k0 += 32) {
    if (tid < 32) {                                  // wave 0: TDM A tile
      tdm_load_2d_bf16(&As[0][0], act + (size_t)base * I_DIM + k0,
                       /*tile*/ 32, 64,
                       /*tensor*/ (unsigned)I_DIM, (unsigned)rows,
                       /*stride*/ (unsigned)I_DIM,
                       /*pad: 16B every 64B*/ 1, 3, 3);
    }
#pragma unroll
    for (int j = 0; j < 8; ++j) {
      int idx = tid + j * 256;
      int n = idx >> 5, k = idx & 31;
      Bs[n][k] = (bf16)W[(size_t)(n0 + n) * I_DIM + k0 + k];
    }
    __builtin_prefetch(&W[(size_t)n0 * I_DIM + k0 + 128], 0, 1);
    wait_tensor0();
    __syncthreads();
    bf16x16 a = frag_a_ld(&As[mw][0], 40);
    c0 = wmma_bf16(a, frag_b_wt(&Bs[nw][0], 40), c0);
    c1 = wmma_bf16(a, frag_b_wt(&Bs[nw + 16][0], 40), c1);
    __syncthreads();
  }
  int l = lane_id(), nl = l & 15, g = l >> 4;
#pragma unroll
  for (int r = 0; r < 8; ++r) {
    int rr = mw + g * 8 + r;
    if (rr < rows) {
      int slot = base + rr;
      int tok = slot_token[slot];
      float wgt = slot_w[slot];
      int n = n0 + nw + nl;
      float y0 = c0[r] + bd[(size_t)e * H_DIM + n];
      float y1 = c1[r] + bd[(size_t)e * H_DIM + n + 16];
      atomicAdd(&out[(size_t)tok * H_DIM + n],      wgt * y0);
      atomicAdd(&out[(size_t)tok * H_DIM + n + 16], wgt * y1);
    }
  }
}

// ---------------------------------------------------------------------------
extern "C" void kernel_launch(void* const* d_in, const int* in_sizes, int n_in,
                              void* d_out, int out_size, void* d_ws, size_t ws_size,
                              hipStream_t stream) {
  (void)in_sizes; (void)n_in; (void)out_size; (void)ws_size;
  const float* hs        = (const float*)d_in[0];
  const float* rms1_w    = (const float*)d_in[1];
  const float* rms2_w    = (const float*)d_in[2];
  const float* w_qkv     = (const float*)d_in[3];
  const float* b_qkv     = (const float*)d_in[4];
  const float* w_o       = (const float*)d_in[5];
  const float* b_o       = (const float*)d_in[6];
  const float* sinks     = (const float*)d_in[7];
  const float* router_w  = (const float*)d_in[8];
  const float* router_b  = (const float*)d_in[9];
  const float* w_gate_up = (const float*)d_in[10];
  const float* b_gate_up = (const float*)d_in[11];
  const float* w_down    = (const float*)d_in[12];
  const float* b_down    = (const float*)d_in[13];
  const int*   positions = (const int*)d_in[14];
  float* out = (float*)d_out;

  char* ws = (char*)d_ws;
  size_t off = 0;
  auto alloc = [&](size_t bytes) -> char* {
    char* p = ws + off;
    off = (off + bytes + 255) & ~(size_t)255;
    return p;
  };
  bf16*  xn         = (bf16*) alloc((size_t)T_TOK * H_DIM * 2);
  float* qkv        = (float*)alloc((size_t)T_TOK * QKV_DIM * 4);
  bf16*  qbf        = (bf16*) alloc((size_t)NHEAD * T_TOK * 64 * 2);
  bf16*  kbf        = (bf16*) alloc((size_t)NKVH * T_TOK * 64 * 2);
  bf16*  vbf        = (bf16*) alloc((size_t)NKVH * T_TOK * 64 * 2);
  bf16*  attn_bf    = (bf16*) alloc((size_t)T_TOK * O_DIM * 2);
  float* x          = (float*)alloc((size_t)T_TOK * H_DIM * 4);
  bf16*  h2         = (bf16*) alloc((size_t)T_TOK * H_DIM * 2);
  int*   topi       = (int*)  alloc((size_t)T_TOK * K_TOP * 4);
  float* topw       = (float*)alloc((size_t)T_TOK * K_TOP * 4);
  int*   counts     = (int*)  alloc(E_EXP * 4);
  int*   offsets    = (int*)  alloc(E_EXP * 4);
  int*   fill       = (int*)  alloc(E_EXP * 4);
  int*   slot_token = (int*)  alloc((size_t)T_TOK * K_TOP * 4);
  float* slot_w     = (float*)alloc((size_t)T_TOK * K_TOP * 4);
  bf16*  act        = (bf16*) alloc((size_t)T_TOK * K_TOP * I_DIM * 2);

  hipMemsetAsync(counts, 0, E_EXP * sizeof(int), stream);

  rmsnorm_kernel<<<T_TOK, 256, 0, stream>>>(hs, rms1_w, xn);
  gemm_xwT_kernel<<<dim3(QKV_DIM / 64, T_TOK / 128), 256, 0, stream>>>(
      xn, w_qkv, b_qkv, nullptr, qkv, T_TOK, QKV_DIM, H_DIM);
  rope_kernel<<<T_TOK, 256, 0, stream>>>(qkv, positions, qbf, kbf, vbf);
  attn_kernel<<<dim3(T_TOK / 16, NHEAD), 32, 0, stream>>>(qbf, kbf, vbf, sinks, attn_bf);
  gemm_xwT_kernel<<<dim3(H_DIM / 64, T_TOK / 128), 256, 0, stream>>>(
      attn_bf, w_o, b_o, hs, x, T_TOK, H_DIM, O_DIM);
  rmsnorm_kernel<<<T_TOK, 256, 0, stream>>>(x, rms2_w, h2);
  router_kernel<<<T_TOK, 32, 0, stream>>>(h2, router_w, router_b, topi, topw, counts);
  moe_scan_kernel<<<1, 32, 0, stream>>>(counts, offsets, fill);
  moe_scatter_kernel<<<(T_TOK + 255) / 256, 256, 0, stream>>>(
      topi, topw, offsets, fill, slot_token, slot_w);
  copy_kernel<<<((T_TOK * H_DIM) + 255) / 256, 256, 0, stream>>>(x, out, T_TOK * H_DIM);
  moe_gateup_kernel<<<dim3(2 * I_DIM / 64, T_TOK / 64, E_EXP), 256, 0, stream>>>(
      h2, w_gate_up, b_gate_up, offsets, counts, slot_token, act);
  moe_down_kernel<<<dim3(H_DIM / 64, T_TOK / 64, E_EXP), 256, 0, stream>>>(
      act, w_down, b_down, offsets, counts, slot_token, slot_w, out);
}
